// MaximallyFixedBlock_44976897524210
// MI455X (gfx1250) — matrix operand order repaired and checked by reading, not compiled, not run on hardware
//
#include <hip/hip_runtime.h>
#include <math.h>

typedef __attribute__((ext_vector_type(16))) __bf16 v16bf;
typedef __attribute__((ext_vector_type(8)))  __bf16 v8bf;
typedef __attribute__((ext_vector_type(8)))  float  v8f;

#define PI_F 3.14159265358979f

static __device__ inline __bf16 to_bf16(float f) {
    union { float f; unsigned u; } v; v.f = f;
    unsigned r = v.u + 0x7FFFu + ((v.u >> 16) & 1u);   // round-to-nearest-even
    unsigned short h = (unsigned short)(r >> 16);
    return __builtin_bit_cast(__bf16, h);
}

// ---------------------------------------------------------------------------
// f32 -> bf16 elementwise convert
// ---------------------------------------------------------------------------
__global__ void k_f32_to_bf16(const float* __restrict__ in, __bf16* __restrict__ out, int n) {
    int i = blockIdx.x * blockDim.x + threadIdx.x;
    if (i < n) out[i] = to_bf16(in[i]);
}

// ---------------------------------------------------------------------------
// Transpose (K x N, leading dim ld) f32 weight into (N x K) bf16
// ---------------------------------------------------------------------------
__global__ void k_transpose_bf16(const float* __restrict__ src, int ld, int K, int N,
                                 __bf16* __restrict__ dst) {
    int i = blockIdx.x * blockDim.x + threadIdx.x;
    if (i >= N * K) return;
    int n = i % N, k = i / N;
    dst[(size_t)n * K + k] = to_bf16(src[(size_t)k * ld + n]);
}

// ---------------------------------------------------------------------------
// WMMA bf16 GEMM, 4 M-tiles per wave (B fragment reused across 4 WMMAs).
// C(MxN) = A(MxK row-major bf16) @ Bt(NxK bf16 = B^T) [+ res]
// M must be a multiple of 64, N of 16, K of 32.
// ---------------------------------------------------------------------------
__global__ void k_gemm_wmma_bf16(const __bf16* __restrict__ A, const __bf16* __restrict__ Bt,
                                 const float* __restrict__ res, float* __restrict__ C,
                                 int M, int N, int K) {
    int wid  = threadIdx.x >> 5;
    int lane = threadIdx.x & 31;
    int Mg = M >> 6, Nt = N >> 4;               // groups of 4 M-tiles
    int tile = blockIdx.x * 8 + wid;
    if (tile >= Mg * Nt) return;
    int tg = tile / Nt, tn = tile % Nt;
    int row = lane & 15, hi = lane >> 4;

    const __bf16* ap = A  + (size_t)(tg * 64 + row) * K + hi * 16;
    const __bf16* bp = Bt + (size_t)(tn * 16 + row) * K + hi * 16;

    v8f acc[4] = {};
    for (int kk = 0; kk < K; kk += 32) {
        v16bf bv = *(const v16bf*)(bp + kk);
#pragma unroll
        for (int m = 0; m < 4; ++m) {
            const __bf16* a = ap + (size_t)(m * 16) * K + kk;
            v8bf a0 = *(const v8bf*)(a);
            v8bf a1 = *(const v8bf*)(a + 8);
            v16bf av;
#pragma unroll
            for (int j = 0; j < 8; ++j) { av[j] = a0[j]; av[j + 8] = a1[j]; }
            acc[m] = __builtin_amdgcn_wmma_f32_16x16x32_bf16(false, av, false, bv,
                                                             (short)0, acc[m], false, false);
        }
    }
    int col = lane & 15;
#pragma unroll
    for (int m = 0; m < 4; ++m) {
#pragma unroll
        for (int r = 0; r < 8; ++r) {
            int rr = tg * 64 + m * 16 + r + hi * 8;
            size_t o = (size_t)rr * N + tn * 16 + col;
            float v = acc[m][r];
            if (res) v += res[o];
            C[o] = v;
        }
    }
}

// ---------------------------------------------------------------------------
// Precompute phasors for every (b,l).
// PH[row*80 + .]: [0:16) kph_re  [16:32) kph_im  [32:48) qph_re [48:64) qph_im
//                 [64:68) jk_re  [68:72) jk_im   [72:76) jq_re  [76:80) jq_im
// PPH[l*32 + .]:  [0:16) pos cos, [16:32) pos sin   (b-independent)
// P layout per row: [0:16) key pre-act, [16:32) query pre-act,
//                   [32:48) x@ltmW1, [48:64) x@ltmW2
// ---------------------------------------------------------------------------
__global__ void k_phasors(const float* __restrict__ P, const float* __restrict__ freqs,
                          float* __restrict__ PH, float* __restrict__ PPH) {
    const int L = 1024;
    int row = blockIdx.x * blockDim.x + threadIdx.x;
    if (row >= 2 * L) return;
    float kr[16], ki[16], qr[16], qi[16];
#pragma unroll
    for (int i = 0; i < 16; ++i) {
        float a = tanhf(P[(size_t)row * 64 + i]) * PI_F;
        __sincosf(a, &ki[i], &kr[i]);
        a = tanhf(P[(size_t)row * 64 + 16 + i]) * PI_F;
        __sincosf(a, &qi[i], &qr[i]);
    }
    float* ph = PH + (size_t)row * 80;
#pragma unroll
    for (int i = 0; i < 16; ++i) {
        ph[i] = kr[i]; ph[16 + i] = ki[i];
        ph[32 + i] = qr[i]; ph[48 + i] = qi[i];
    }
#pragma unroll
    for (int p = 0; p < 4; ++p) {
        float r = 1.f, im = 0.f;
#pragma unroll
        for (int s = 0; s < 4; ++s) {
            int i = s * 4 + p;
            float nr = r * kr[i] - im * ki[i];
            im = r * ki[i] + im * kr[i];
            r = nr;
        }
        ph[64 + p] = r; ph[68 + p] = im;
        r = 1.f; im = 0.f;
#pragma unroll
        for (int s = 0; s < 4; ++s) {
            int i = s * 4 + p;
            float nr = r * qr[i] - im * qi[i];
            im = r * qi[i] + im * qr[i];
            r = nr;
        }
        ph[72 + p] = r; ph[76 + p] = im;
    }
    if (row < L) {
#pragma unroll
        for (int j = 0; j < 16; ++j) {
            float s, c;
            __sincosf(2.f * PI_F * freqs[j] * (float)row, &s, &c);
            PPH[(size_t)row * 32 + j] = c;
            PPH[(size_t)row * 32 + 16 + j] = s;
        }
    }
}

// ---------------------------------------------------------------------------
// Tiny serial scan for the write gate: joint-key memory is only 4 complex
// values per step and is independent of V. One wave per batch; lanes 0..3
// own the 4 planes, magnitude reduced with wave32 shuffles.
// ---------------------------------------------------------------------------
__global__ void k_gates(const float* __restrict__ PH, float* __restrict__ gate) {
    const int L = 1024;
    int b = blockIdx.x;
    int lane = threadIdx.x;
    float jm_re = 0.f, jm_im = 0.f;
    for (int l = 0; l < L; ++l) {
        size_t row = (size_t)b * L + l;
        float mag4 = 0.f, jkre = 0.f, jkim = 0.f;
        if (lane < 4) {
            jkre = PH[row * 80 + 64 + lane];
            jkim = PH[row * 80 + 68 + lane];
            float jqre = PH[row * 80 + 72 + lane];
            float jqim = PH[row * 80 + 76 + lane];
            float rr = jm_re * jqre + jm_im * jqim;
            float ri = jm_im * jqre - jm_re * jqim;
            mag4 = sqrtf(rr * rr + ri * ri);
        }
        mag4 += __shfl_xor(mag4, 1, 32);
        mag4 += __shfl_xor(mag4, 2, 32);
        float mag = __shfl(mag4, 0, 32) * 0.25f;
        if (lane == 0) {
            float posn = fmaxf((float)l, 1.f);
            float nres = mag * __frsqrt_rn(posn);
            float sur = 0.5f * (1.f - tanhf(5.f * (nres - 0.3f)));
            gate[row] = 1.f / (1.f + __expf(-5.f * (sur - 0.5f)));
        }
        if (lane < 4) { jm_re += jkre; jm_im += jkim; }
    }
}

// CS component layout (stride S = B*NC*256 floats per component):
//   0..7   mc_re[4], mc_im[4]       (cross-bank)
//   8..39  ma_re[16], ma_im[16]     (per-bank)
//   40..71 mp_re[16], mp_im[16]     (positional)
//   72,73  s1, s2                   (running sums of x, x^2)
#define NCOMP 74

// ---------------------------------------------------------------------------
// Pass 1: per-chunk totals of every prefix-summed quantity.
// Grid: B*NC blocks, 256 threads (thread = channel d).
// ---------------------------------------------------------------------------
__global__ void k_chunksum(const float* __restrict__ x, const float* __restrict__ V,
                           const float* __restrict__ P, const float* __restrict__ PH,
                           const float* __restrict__ PPH, const float* __restrict__ gate,
                           float* __restrict__ CS, float* __restrict__ CU) {
    const int L = 1024, D = 256, NC = 16, CL = 64;
    int bc = blockIdx.x;
    int b = bc / NC, c = bc % NC;
    int t = threadIdx.x;
    __shared__ float ph[80];
    __shared__ float pp[32];
    float mc[8] = {}, ma[32] = {}, mp[32] = {};
    float s1 = 0.f, s2 = 0.f, cu = 0.f;

    for (int i = 0; i < CL; ++i) {
        int l = c * CL + i;
        size_t row = (size_t)b * L + l;
        if (t < 80) ph[t] = PH[row * 80 + t];
        if (t >= 80 && t < 112) pp[t - 80] = PPH[(size_t)l * 32 + (t - 80)];
        float xd = x[row * D + t];
        if (t < 16) cu += P[row * 64 + 48 + t];
        s1 += xd; s2 += xd * xd;
        __syncthreads();
        float Vd = V[row * D + t];
        float Vg = Vd * gate[row];
#pragma unroll
        for (int p = 0; p < 4; ++p) {
            mc[p]     += ph[64 + p] * Vg;
            mc[4 + p] += ph[68 + p] * Vg;
        }
#pragma unroll
        for (int j = 0; j < 16; ++j) {
            ma[j]      += ph[j] * Vg;
            ma[16 + j] += ph[16 + j] * Vg;
            mp[j]      += pp[j] * Vd;
            mp[16 + j] += pp[16 + j] * Vd;
        }
        __syncthreads();
    }
    const int S = 2 * NC * 256;
    size_t base = (size_t)bc * 256 + t;
#pragma unroll
    for (int p = 0; p < 8; ++p)  CS[(size_t)p * S + base] = mc[p];
#pragma unroll
    for (int j = 0; j < 32; ++j) CS[(size_t)(8 + j) * S + base] = ma[j];
#pragma unroll
    for (int j = 0; j < 32; ++j) CS[(size_t)(40 + j) * S + base] = mp[j];
    CS[(size_t)72 * S + base] = s1;
    CS[(size_t)73 * S + base] = s2;
    if (t < 16) CU[bc * 16 + t] = cu;
}

// ---------------------------------------------------------------------------
// Pass 2: in-place exclusive scan of the chunk totals (over the 16 chunks).
// ---------------------------------------------------------------------------
__global__ void k_chunkscan(float* __restrict__ CS, float* __restrict__ CU) {
    const int NC = 16;
    int b = blockIdx.x;
    int t = threadIdx.x;
    const int S = 2 * NC * 256;
    for (int comp = 0; comp < NCOMP; ++comp) {
        float run = 0.f;
        for (int c = 0; c < NC; ++c) {
            size_t idx = (size_t)comp * S + (size_t)(b * NC + c) * 256 + t;
            float v = CS[idx];
            CS[idx] = run;
            run += v;
        }
    }
    if (t < 16) {
        float run = 0.f;
        for (int c = 0; c < NC; ++c) {
            int idx = (b * NC + c) * 16 + t;
            float v = CU[idx];
            CU[idx] = run;
            run += v;
        }
    }
}

// ---------------------------------------------------------------------------
// Pass 3: chunk-local scan seeded with exclusive offsets; does retrieval,
// the nonlinear running-std/LTM path, and writes the normalized `pre`.
// ---------------------------------------------------------------------------
__global__ void k_chunkretrieve(const float* __restrict__ x, const float* __restrict__ V,
                                const float* __restrict__ P, const float* __restrict__ PH,
                                const float* __restrict__ PPH, const float* __restrict__ gate,
                                const float* __restrict__ CS, const float* __restrict__ CU,
                                const float* __restrict__ ltm_key_proj,
                                const float* __restrict__ bmre, const float* __restrict__ bmim,
                                float* __restrict__ pre) {
    const int L = 1024, D = 256, NC = 16, CL = 64;
    int bc = blockIdx.x;
    int b = bc / NC, c = bc % NC;
    int t = threadIdx.x, lane = t & 31, wid = t >> 5;

    __shared__ float ph[80];
    __shared__ float pp[32];
    __shared__ float sL[32];
    __shared__ float lc[16], ls[16];
    __shared__ float red[8][16];

    const int S = 2 * NC * 256;
    size_t base = (size_t)bc * 256 + t;
    float mc[8], ma[32], mp[32];
#pragma unroll
    for (int p = 0; p < 8; ++p)  mc[p] = CS[(size_t)p * S + base];
#pragma unroll
    for (int j = 0; j < 32; ++j) ma[j] = CS[(size_t)(8 + j) * S + base];
#pragma unroll
    for (int j = 0; j < 32; ++j) mp[j] = CS[(size_t)(40 + j) * S + base];
    float s1 = CS[(size_t)72 * S + base];
    float s2 = CS[(size_t)73 * S + base];
    float cs2 = (t >= 32 && t < 48) ? CU[bc * 16 + (t - 32)] : 0.f;

    float w3[16], bre[16], bim[16];
#pragma unroll
    for (int j = 0; j < 16; ++j) {
        w3[j]  = ltm_key_proj[(size_t)(512 + t) * 16 + j];
        bre[j] = bmre[j * 256 + t];
        bim[j] = bmim[j * 256 + t];
    }

    for (int i = 0; i < CL; ++i) {
        int l = c * CL + i;
        size_t row = (size_t)b * L + l;
        float p1 = (float)(l + 1);

        // ---- phase A: stage uniforms, update running stats, reduce std@W3
        if (t < 80) ph[t] = PH[row * 80 + t];
        if (t >= 80 && t < 112) pp[t - 80] = PPH[(size_t)l * 32 + (t - 80)];
        if (t >= 112 && t < 144) sL[t - 112] = P[row * 64 + 32 + (t - 112)];
        float xd = x[row * D + t];
        s1 += xd; s2 += xd * xd;
        float mean = s1 / p1;
        float var  = s2 / p1 - mean * mean;
        float sd   = sqrtf(fmaxf(var, 1e-8f));
#pragma unroll
        for (int j = 0; j < 16; ++j) {
            float v = sd * w3[j];
#pragma unroll
            for (int o = 16; o; o >>= 1) v += __shfl_xor(v, o, 32);
            if (lane == 0) red[wid][j] = v;
        }
        __syncthreads();

        // ---- phase B: LTM phasor
        if (t >= 32 && t < 48) {
            int j = t - 32;
            float l3 = 0.f;
#pragma unroll
            for (int w = 0; w < 8; ++w) l3 += red[w][j];
            cs2 += sL[16 + j];
            float lv = sL[j] + cs2 / p1 + l3;
            float a = tanhf(lv) * PI_F;
            float s, cc; __sincosf(a, &s, &cc);
            lc[j] = cc; ls[j] = s;
        }
        __syncthreads();

        // ---- phase C: memory updates + retrieval
        float Vd = V[row * D + t];
        float Vg = Vd * gate[row];

        float cross = 0.f;
#pragma unroll
        for (int p = 0; p < 4; ++p) {
            mc[p]     += ph[64 + p] * Vg;
            mc[4 + p] += ph[68 + p] * Vg;
            cross += mc[p] * ph[72 + p] + mc[4 + p] * ph[76 + p];
        }
        float banks = 0.f;
#pragma unroll
        for (int j = 0; j < 16; ++j) {
            ma[j]      += ph[j] * Vg;
            ma[16 + j] += ph[16 + j] * Vg;
            banks += ma[j] * ph[32 + j] + ma[16 + j] * ph[48 + j];
        }
        banks *= 0.25f;
        float rpos = 0.f;
#pragma unroll
        for (int j = 0; j < 16; ++j) {
            mp[j]      += pp[j] * Vd;
            mp[16 + j] += pp[16 + j] * Vd;
            rpos += mp[j] * pp[j] + mp[16 + j] * pp[16 + j];
        }
        float pers = 0.f;
#pragma unroll
        for (int j = 0; j < 16; ++j) pers += bre[j] * lc[j] + bim[j] * ls[j];
        pers *= 0.25f;

        float total = 0.2f * (banks + cross);
        total = 0.5f * total + 0.5f * rpos + 0.5f * pers;
        pre[row * D + t] = total * __frsqrt_rn(p1 * 4.f);
        __syncthreads();
    }
}

// ---------------------------------------------------------------------------
extern "C" void kernel_launch(void* const* d_in, const int* in_sizes, int n_in,
                              void* d_out, int out_size, void* d_ws, size_t ws_size,
                              hipStream_t stream) {
    const float* x            = (const float*)d_in[0];
    const float* key_proj     = (const float*)d_in[1];
    const float* query_proj   = (const float*)d_in[2];
    const float* value_proj   = (const float*)d_in[3];
    const float* out_proj     = (const float*)d_in[4];
    const float* ltm_key_proj = (const float*)d_in[5];
    const float* pos_freqs    = (const float*)d_in[7];
    const float* bmre         = (const float*)d_in[10];
    const float* bmim         = (const float*)d_in[11];
    float* out = (float*)d_out;

    const int B = 2, L = 1024, D = 256, NC = 16;
    const int M = B * L;
    const int MK = M * D;

    char* ws = (char*)d_ws;
    __bf16* x_bf   = (__bf16*)(ws + 0);           // 1 MB
    __bf16* Wv_t   = (__bf16*)(ws + 1048576);     // 128 KB
    __bf16* Wo_t   = (__bf16*)(ws + 1179648);     // 128 KB
    __bf16* Wkq_t  = (__bf16*)(ws + 1310720);     // 32 KB
    float*  Vb     = (float*) (ws + 1343488);     // 2 MB
    float*  Pm     = (float*) (ws + 3440640);     // 512 KB
    float*  pre    = (float*) (ws + 3964928);     // 2 MB
    __bf16* pre_bf = (__bf16*)(ws + 6062080);     // 1 MB
    float*  PH     = (float*) (ws + 7110656);     // 640 KB  (2048 x 80)
    float*  PPH    = (float*) (ws + 7766016);     // 128 KB  (1024 x 32)
    float*  gate   = (float*) (ws + 7897088);     // 8 KB
    float*  CS     = (float*) (ws + 7905280);     // ~2.4 MB (74 x 32 x 256)
    float*  CU     = (float*) (ws + 10330112);    // 2 KB

    k_f32_to_bf16<<<(MK + 255) / 256, 256, 0, stream>>>(x, x_bf, MK);

    k_transpose_bf16<<<(256 * 256 + 255) / 256, 256, 0, stream>>>(value_proj, 256, 256, 256, Wv_t);
    k_transpose_bf16<<<(256 * 256 + 255) / 256, 256, 0, stream>>>(out_proj,   256, 256, 256, Wo_t);
    k_transpose_bf16<<<(16 * 256 + 255) / 256, 256, 0, stream>>>(key_proj,   16, 256, 16, Wkq_t + 0  * 256);
    k_transpose_bf16<<<(16 * 256 + 255) / 256, 256, 0, stream>>>(query_proj, 16, 256, 16, Wkq_t + 16 * 256);
    k_transpose_bf16<<<(16 * 256 + 255) / 256, 256, 0, stream>>>(ltm_key_proj,            16, 256, 16, Wkq_t + 32 * 256);
    k_transpose_bf16<<<(16 * 256 + 255) / 256, 256, 0, stream>>>(ltm_key_proj + 256 * 16, 16, 256, 16, Wkq_t + 48 * 256);

    // V = x @ value_proj   (2048 x 256 x 256)
    int tilesV = (M / 64) * (256 / 16);
    k_gemm_wmma_bf16<<<(tilesV + 7) / 8, 256, 0, stream>>>(x_bf, Wv_t, nullptr, Vb, M, 256, 256);
    // P = x @ [key|query|ltmW1|ltmW2]   (2048 x 64 x 256)
    int tilesP = (M / 64) * (64 / 16);
    k_gemm_wmma_bf16<<<(tilesP + 7) / 8, 256, 0, stream>>>(x_bf, Wkq_t, nullptr, Pm, M, 64, 256);

    // phasors + gate scan + chunked parallel prefix scan
    k_phasors<<<(2 * L + 255) / 256, 256, 0, stream>>>(Pm, pos_freqs, PH, PPH);
    k_gates<<<B, 32, 0, stream>>>(PH, gate);
    k_chunksum<<<B * NC, 256, 0, stream>>>(x, Vb, Pm, PH, PPH, gate, CS, CU);
    k_chunkscan<<<B, 256, 0, stream>>>(CS, CU);
    k_chunkretrieve<<<B * NC, 256, 0, stream>>>(x, Vb, Pm, PH, PPH, gate, CS, CU,
                                                ltm_key_proj, bmre, bmim, pre);

    // out = x + pre @ out_proj
    k_f32_to_bf16<<<(MK + 255) / 256, 256, 0, stream>>>(pre, pre_bf, MK);
    k_gemm_wmma_bf16<<<(tilesV + 7) / 8, 256, 0, stream>>>(pre_bf, Wo_t, x, out, M, 256, 256);
}
